// RNN_53214644797476
// MI455X (gfx1250) — matrix operand order repaired and checked by reading, not compiled, not run on hardware
//
#include <hip/hip_runtime.h>
#include <hip/hip_bf16.h>
#include <math.h>

#define SEQ   512
#define BATCH 64
#define INF_  1024   // IN
#define HIDN  1024   // HID

typedef __attribute__((ext_vector_type(16))) __bf16        v16bf;
typedef __attribute__((ext_vector_type(2)))  __bf16        v2bf;
typedef __attribute__((ext_vector_type(8)))  float         v8f;
typedef __attribute__((ext_vector_type(8)))  unsigned int  v8u;
typedef __attribute__((ext_vector_type(4)))  unsigned int  u4;
typedef __attribute__((ext_vector_type(2)))  unsigned int  u2;
typedef __attribute__((ext_vector_type(4)))  float         f4;

// Global (addrspace 1) pointer typedefs -> force global_load/global_store
// instead of flat_load/flat_store (no DScnt coupling, SADDR addressing).
typedef const float          __attribute__((address_space(1))) * gf_p;
typedef const f4             __attribute__((address_space(1))) * gf4_p;
typedef const u4             __attribute__((address_space(1))) * gu4_p;
typedef const unsigned short __attribute__((address_space(1))) * gus_p;
typedef float                __attribute__((address_space(1))) * gfw_p;

__device__ __forceinline__ gf_p  as_gf (const void* p) { return (gf_p) (unsigned long long)p; }
__device__ __forceinline__ gus_p as_gus(const void* p) { return (gus_p)(unsigned long long)p; }
__device__ __forceinline__ gfw_p as_gfw(void* p)       { return (gfw_p)(unsigned long long)p; }

// Pack two f32 -> one dword of two bf16 via the CDNA5 single-instruction path.
__device__ __forceinline__ unsigned int pkbf(float lo, float hi) {
#if __has_builtin(__builtin_amdgcn_cvt_pk_bf16_f32)
    v2bf r = __builtin_amdgcn_cvt_pk_bf16_f32(lo, hi);
    return __builtin_bit_cast(unsigned int, r);
#else
    unsigned int d;
    asm("v_cvt_pk_bf16_f32 %0, %1, %2" : "=v"(d) : "v"(lo), "v"(hi));
    return d;
#endif
}

// Hardware tanh (TRANS op on CDNA5).
__device__ __forceinline__ float fast_tanh(float x) {
#if __has_builtin(__builtin_amdgcn_tanhf)
    return __builtin_amdgcn_tanhf(x);
#else
    float r;
    asm("v_tanh_f32 %0, %1" : "=v"(r) : "v"(x));
    return r;
#endif
}

// ---------------------------------------------------------------------------
// h_term[b,h] = sum_i hidden[b,i] * H[h,i]   (64 x 1024, tiny: plain f32)
// ---------------------------------------------------------------------------
__global__ __launch_bounds__(256) void hterm_kernel(
    const float* __restrict__ hs, const float* __restrict__ Hm,
    float* __restrict__ ht) {
    int idx = blockIdx.x * 256 + threadIdx.x;          // 64*1024 threads
    int b = idx >> 10;
    int h = idx & (HIDN - 1);
    gf4_p a = (gf4_p)as_gf(hs + (size_t)b * HIDN);
    gf4_p w = (gf4_p)as_gf(Hm + (size_t)h * HIDN);
    float s = 0.f;
    for (int i = 0; i < HIDN / 4; ++i) {
        f4 x = a[i], y = w[i];
        s += x.x * y.x + x.y * y.y + x.z * y.z + x.w * y.w;
    }
    as_gfw(ht)[idx] = s;
}

// ---------------------------------------------------------------------------
// One-time W (f32, [HID][IN]) -> bf16 copy in workspace (W is L2-resident;
// removes all B-side conversion from the GEMM inner loop).
// ---------------------------------------------------------------------------
__global__ __launch_bounds__(256) void wconv_kernel(
    const float* __restrict__ W, unsigned int* __restrict__ Wb) {
    int idx = blockIdx.x * 256 + threadIdx.x;          // HID*IN/4 threads
    f4 v = ((gf4_p)as_gf(W))[idx];
    u2 r;
    r.x = pkbf(v.x, v.y);
    r.y = pkbf(v.z, v.w);
    ((u2 __attribute__((address_space(1)))*)as_gfw((void*)Wb))[idx] = r;
}

// ---------------------------------------------------------------------------
// Main fused GEMM: out[m,n] = tanh( X[m,:] . W[n,:] + ht[m&63, n] )
//   M = SEQ*BATCH = 32768, N = HID = 1024, K = IN = 1024
// Block tile 128x128, 8 waves (wave32), wave tile 32x64 = 2x4 WMMA accums.
// A: f32 stream + inline v_cvt_pk_bf16_f32.  B: pre-converted bf16 (L2).
// ---------------------------------------------------------------------------
__global__ __launch_bounds__(256) void rnn_gemm_kernel(
    const float* __restrict__ X, const unsigned short* __restrict__ Wbv,
    const float* __restrict__ htv, float* __restrict__ outv) {

    gf_p  Xg  = as_gf(X);
    gus_p Wg  = as_gus(Wbv);
    gf_p  htg = as_gf(htv);
    gfw_p og  = as_gfw(outv);

    const int lane  = threadIdx.x & 31;
    const int wave  = threadIdx.x >> 5;
    const int waveM = wave >> 1;          // 0..3
    const int waveN = wave & 1;           // 0..1
    const int half  = lane >> 4;          // 0 or 1
    const int l15   = lane & 15;

    const int m0 = blockIdx.y * 128 + waveM * 32;
    const int n0 = blockIdx.x * 128 + waveN * 64;

    // Per-lane 32-bit element offsets (loop-invariant); k0 is loop-uniform.
    // A (16-bit 16x32 layout): lanes 0-15 hold K {0..7,16..23} of row M=l15,
    // lanes 16-31 hold K {8..15,24..31}.
    unsigned int aoff[2];
#pragma unroll
    for (int tm = 0; tm < 2; ++tm)
        aoff[tm] = (unsigned)(m0 + tm * 16 + l15) * INF_ + half * 8;

    // B (16-bit 32x16 layout): lane holds column n (a row of W); lanes 0-15
    // hold K {0..7,16..23}, lanes 16-31 hold K {8..15,24..31}.
    unsigned int boff[4];
#pragma unroll
    for (int tn = 0; tn < 4; ++tn)
        boff[tn] = (unsigned)(n0 + tn * 16 + l15) * INF_ + half * 8;

    v8f acc[2][4] = {};

#pragma unroll 2
    for (int k0 = 0; k0 < INF_; k0 += 32) {
        v16bf afrag[2], bfrag[4];

#pragma unroll
        for (int tm = 0; tm < 2; ++tm) {
            gf4_p p = (gf4_p)(Xg + aoff[tm] + k0);
            f4 c0 = p[0];     // K+0..3
            f4 c1 = p[1];     // K+4..7
            f4 c2 = p[4];     // K+16..19
            f4 c3 = p[5];     // K+20..23
            v8u u;
            u[0] = pkbf(c0.x, c0.y); u[1] = pkbf(c0.z, c0.w);
            u[2] = pkbf(c1.x, c1.y); u[3] = pkbf(c1.z, c1.w);
            u[4] = pkbf(c2.x, c2.y); u[5] = pkbf(c2.z, c2.w);
            u[6] = pkbf(c3.x, c3.y); u[7] = pkbf(c3.z, c3.w);
            afrag[tm] = __builtin_bit_cast(v16bf, u);
        }

#pragma unroll
        for (int tn = 0; tn < 4; ++tn) {
            gu4_p p = (gu4_p)(Wg + boff[tn] + k0);
            u4 p0 = p[0];     // bf16 K+0..7
            u4 p1 = p[2];     // bf16 K+16..23
            v8u u = __builtin_shufflevector(p0, p1, 0, 1, 2, 3, 4, 5, 6, 7);
            bfrag[tn] = __builtin_bit_cast(v16bf, u);
        }

#pragma unroll
        for (int tm = 0; tm < 2; ++tm)
#pragma unroll
            for (int tn = 0; tn < 4; ++tn)
                acc[tm][tn] = __builtin_amdgcn_wmma_f32_16x16x32_bf16(
                    false, afrag[tm], false, bfrag[tn],
                    (short)0, acc[tm][tn], false, false);
    }

    // Epilogue: + h_term[b,n], tanh, store; duplicate t == SEQ-1 rows to tail.
#pragma unroll
    for (int tm = 0; tm < 2; ++tm) {
#pragma unroll
        for (int tn = 0; tn < 4; ++tn) {
            const int col  = n0 + tn * 16 + l15;
            const int rowb = m0 + tm * 16 + half * 8;
#pragma unroll
            for (int r = 0; r < 8; ++r) {
                const int row = rowb + r;                 // m = t*BATCH + b
                float v = acc[tm][tn][r] + htg[(row & (BATCH - 1)) * HIDN + col];
                v = fast_tanh(v);
                og[(size_t)row * HIDN + col] = v;
                if (row >= (SEQ - 1) * BATCH)             // last timestep copy
                    og[(size_t)SEQ * BATCH * HIDN +
                       (size_t)(row - (SEQ - 1) * BATCH) * HIDN + col] = v;
            }
        }
    }
}

// ---------------------------------------------------------------------------
extern "C" void kernel_launch(void* const* d_in, const int* in_sizes, int n_in,
                              void* d_out, int out_size, void* d_ws, size_t ws_size,
                              hipStream_t stream) {
    const float* X  = (const float*)d_in[0];   // [SEQ, BATCH, IN]
    const float* hs = (const float*)d_in[1];   // [BATCH, HID]
    const float* W  = (const float*)d_in[2];   // [HID, IN]
    const float* Hm = (const float*)d_in[3];   // [HID, HID]
    float*       out = (float*)d_out;          // [SEQ*BATCH*HID + BATCH*HID]

    // Workspace layout: [0, 256KB) h_term f32; [256KB, 256KB+2MB) W in bf16.
    float*          ht = (float*)d_ws;
    unsigned int*   Wb = (unsigned int*)((char*)d_ws + (size_t)BATCH * HIDN * 4);

    hterm_kernel<<<(BATCH * HIDN) / 256, 256, 0, stream>>>(hs, Hm, ht);
    wconv_kernel<<<(HIDN * INF_ / 4) / 256, 256, 0, stream>>>(W, Wb);

    dim3 grid(HIDN / 128, (SEQ * BATCH) / 128);
    rnn_gemm_kernel<<<grid, 256, 0, stream>>>(X, (const unsigned short*)Wb, ht, out);
}
